// SpatiallyAwareTransformer_45140106281256
// MI455X (gfx1250) — matrix operand order repaired and checked by reading, not compiled, not run on hardware
//
#include <hip/hip_runtime.h>
#include <cmath>

typedef _Float16 f16;
typedef __attribute__((ext_vector_type(16))) _Float16 v16h;
typedef __attribute__((ext_vector_type(8)))  _Float16 v8h;
typedef __attribute__((ext_vector_type(8)))  float    v8f;
typedef __attribute__((ext_vector_type(4)))  float    v4f;

#define NHEADS   8
#define DH       32
#define S_TOT    5440
#define BATCH    4
#define NQ       100
#define MEMPOOL  512

__device__ __forceinline__ int imin(int a, int b) { return a < b ? a : b; }
__device__ __forceinline__ int imax(int a, int b) { return a > b ? a : b; }

__device__ __forceinline__ void level_of(int s, int& lvl, int& row, int& col,
                                         int& H, int& W, int& st) {
  if (s < 4096)      { lvl = 0; st = 0;    H = 64; W = 64; }
  else if (s < 5120) { lvl = 1; st = 4096; H = 32; W = 32; }
  else if (s < 5376) { lvl = 2; st = 5120; H = 16; W = 16; }
  else               { lvl = 3; st = 5376; H = 8;  W = 8;  }
  int r = s - st; row = r / W; col = r % W;
}

__device__ __forceinline__ v16h cat8(v8h a, v8h b) {
  return __builtin_shufflevector(a, b, 0, 1, 2, 3, 4, 5, 6, 7,
                                 8, 9, 10, 11, 12, 13, 14, 15);
}
__device__ __forceinline__ v8h cvt8(v4f x0, v4f x1) {
  v8h h;
  h[0] = (f16)x0[0]; h[1] = (f16)x0[1]; h[2] = (f16)x0[2]; h[3] = (f16)x0[3];
  h[4] = (f16)x1[0]; h[5] = (f16)x1[1]; h[6] = (f16)x1[2]; h[7] = (f16)x1[3];
  return h;
}

// ---- WMMA GEMM: C[M,N] = act(A[M,K] @ W[N,K]^T + bias) --------------------
// Assumptions (hold for all launches here): K % 32 == 0 and N % 128 gives
// exact block coverage (N in {128,256,1024}); only M may be ragged.
// grid: (ceil(M/64), N/128), block: 256 (8 waves, each a 32x32 output tile).
#define LDP 40  // LDS row stride in halfs: 80B, 16B-aligned, bank-spread
__global__ void gemm_wmma_kernel(const float* __restrict__ A,
                                 const float* __restrict__ Wm,
                                 const float* __restrict__ bias,
                                 float* __restrict__ C,
                                 int M, int N, int K, int relu) {
  __shared__ f16 As[64][LDP];
  __shared__ f16 Bs[128][LDP];
  int tid = threadIdx.x;
  int wave = tid >> 5, lane = tid & 31;
  int half = lane >> 4, l15 = lane & 15;
  int m0 = blockIdx.x * 64, n0 = blockIdx.y * 128;
  int wm = (wave >> 2) * 32, wn = (wave & 3) * 32;

  int ar = tid >> 2, asg = (tid & 3) * 8;             // A: row 0..63, 8-col seg
  int amr = imin(m0 + ar, M - 1);
  const float* aptr = A + (size_t)amr * K + asg;
  v8f acc[2][2];
  acc[0][0] = v8f{}; acc[0][1] = v8f{}; acc[1][0] = v8f{}; acc[1][1] = v8f{};

  for (int k0 = 0; k0 < K; k0 += 32) {
    { // ---- stage A (64x32) ----
      const float* s = aptr + k0;
      v4f x0 = *(const v4f*)s;
      v4f x1 = *(const v4f*)(s + 4);
      __builtin_prefetch(s + 32, 0, 0);
      *(v8h*)&As[ar][asg] = cvt8(x0, x1);
    }
    #pragma unroll
    for (int i = 0; i < 2; ++i) { // ---- stage B (128x32) ----
      int idx = tid + i * 256;
      int r = idx >> 2, sg = (idx & 3) * 8;
      const float* s = Wm + (size_t)(n0 + r) * K + k0 + sg;
      v4f x0 = *(const v4f*)s;
      v4f x1 = *(const v4f*)(s + 4);
      __builtin_prefetch(s + 32, 0, 0);
      *(v8h*)&Bs[r][sg] = cvt8(x0, x1);
    }
    __syncthreads();
    v16h afr[2], bfr[2];
    #pragma unroll
    for (int sm = 0; sm < 2; ++sm) {
      int row = wm + sm * 16 + l15;
      afr[sm] = cat8(*(const v8h*)&As[row][8 * half],
                     *(const v8h*)&As[row][16 + 8 * half]);
    }
    #pragma unroll
    for (int sn = 0; sn < 2; ++sn) {
      int col = wn + sn * 16 + l15;
      bfr[sn] = cat8(*(const v8h*)&Bs[col][8 * half],
                     *(const v8h*)&Bs[col][16 + 8 * half]);
    }
    #pragma unroll
    for (int sm = 0; sm < 2; ++sm)
      #pragma unroll
      for (int sn = 0; sn < 2; ++sn)
        acc[sm][sn] = __builtin_amdgcn_wmma_f32_16x16x32_f16(
            false, afr[sm], false, bfr[sn], (short)0, acc[sm][sn], false, false);
    __syncthreads();
  }
  #pragma unroll
  for (int sn = 0; sn < 2; ++sn) {
    int n = n0 + wn + sn * 16 + l15;
    float bv = bias[n];  // N coverage is exact: no guard needed
    #pragma unroll
    for (int sm = 0; sm < 2; ++sm) {
      #pragma unroll
      for (int r = 0; r < 8; ++r) {
        int m = m0 + wm + sm * 16 + r + 8 * half;
        if (m < M) {
          float v = acc[sm][sn][r] + bv;
          if (relu) v = v > 0.f ? v : 0.f;
          C[(size_t)m * N + n] = v;
        }
      }
    }
  }
}

// ---- Cooperative flash attention (dh=32, D=256, 8 heads) ------------------
// Block = 256 threads = 8 waves covering 128 queries; each 32-key K/V tile is
// staged to LDS ONCE per block (8x less K/V cache traffic than per-wave).
// grid: (ceil(Lq/128), 8, B)
__global__ void attn_flash_kernel(const float* __restrict__ Qm,
                                  const float* __restrict__ Km,
                                  const float* __restrict__ Vm,
                                  float* __restrict__ Out, int Lq, int Lk) {
  __shared__ f16 Ks[32][LDP];      // [key][d]
  __shared__ f16 Vs[32][LDP];      // [d][key]  (transposed for P.V B-frags)
  __shared__ f16 Ps[8][16][LDP];   // per-wave P relayout slices
  int tid = threadIdx.x;
  int wave = tid >> 5, lane = tid & 31;
  int half = lane >> 4, l15 = lane & 15;
  int h = blockIdx.y, b = blockIdx.z;
  int q0 = blockIdx.x * 128 + wave * 16;
  const float scale = 0.17677669529663687f; // 1/sqrt(32)
  size_t qbase = (size_t)b * Lq * 256 + h * DH;
  size_t kbase = (size_t)b * Lk * 256 + h * DH;

  v16h qa; // Q as A fragment, scale folded in
  {
    int m = imin(q0 + l15, Lq - 1);
    const float* qrow = Qm + qbase + (size_t)m * 256;
    v4f q0v = *(const v4f*)(qrow + 8 * half);
    v4f q1v = *(const v4f*)(qrow + 8 * half + 4);
    v4f q2v = *(const v4f*)(qrow + 16 + 8 * half);
    v4f q3v = *(const v4f*)(qrow + 16 + 8 * half + 4);
    #pragma unroll
    for (int e = 0; e < 4; ++e) {
      qa[e]      = (f16)(q0v[e] * scale);
      qa[e + 4]  = (f16)(q1v[e] * scale);
      qa[e + 8]  = (f16)(q2v[e] * scale);
      qa[e + 12] = (f16)(q3v[e] * scale);
    }
  }
  int skey = tid >> 3, sd = (tid & 7) * 4;  // staging role: key row, 4-d seg
  float run_m[8], run_l[8];
  v8f acc0 = {}, acc1 = {};
  #pragma unroll
  for (int r = 0; r < 8; ++r) { run_m[r] = -1e30f; run_l[r] = 0.f; }

  for (int j0 = 0; j0 < Lk; j0 += 32) {
    __syncthreads();  // previous tile fully consumed
    { // ---- stage K and V (32 keys x 32 dims), coalesced v4f loads ----
      int j = imin(j0 + skey, Lk - 1);
      const float* krow = Km + kbase + (size_t)j * 256 + sd;
      v4f kx = *(const v4f*)krow;
      f16* kp = &Ks[skey][sd];
      kp[0] = (f16)kx[0]; kp[1] = (f16)kx[1]; kp[2] = (f16)kx[2]; kp[3] = (f16)kx[3];
      const float* vrow = Vm + kbase + (size_t)j * 256 + sd;
      v4f vx = *(const v4f*)vrow;
      Vs[sd + 0][skey] = (f16)vx[0]; Vs[sd + 1][skey] = (f16)vx[1];
      Vs[sd + 2][skey] = (f16)vx[2]; Vs[sd + 3][skey] = (f16)vx[3];
    }
    __syncthreads();
    // ---- scores: Q(16x32) x K^T(32x32) via 2 WMMAs ----
    v8f s[2];
    #pragma unroll
    for (int t = 0; t < 2; ++t) {
      int kn = t * 16 + l15;
      v16h kf = cat8(*(const v8h*)&Ks[kn][8 * half],
                     *(const v8h*)&Ks[kn][16 + 8 * half]);
      v8f z = {};
      s[t] = __builtin_amdgcn_wmma_f32_16x16x32_f16(false, qa, false, kf,
                                                    (short)0, z, false, false);
      if (j0 + kn >= Lk) {
        #pragma unroll
        for (int r = 0; r < 8; ++r) s[t][r] = -1e30f;
      }
    }
    // ---- online softmax (row = r + 8*half, reduced over 16-lane halves) ----
    #pragma unroll
    for (int r = 0; r < 8; ++r) {
      float mx = fmaxf(s[0][r], s[1][r]);
      #pragma unroll
      for (int off = 8; off >= 1; off >>= 1) mx = fmaxf(mx, __shfl_xor(mx, off, 32));
      float newm = fmaxf(run_m[r], mx);
      float corr = __expf(run_m[r] - newm);
      run_m[r] = newm;
      float p0 = __expf(s[0][r] - newm);
      float p1 = __expf(s[1][r] - newm);
      float rs = p0 + p1;
      #pragma unroll
      for (int off = 8; off >= 1; off >>= 1) rs += __shfl_xor(rs, off, 32);
      run_l[r] = run_l[r] * corr + rs;
      acc0[r] *= corr; acc1[r] *= corr;
      s[0][r] = p0; s[1][r] = p1;
    }
    // ---- P: C-layout -> own LDS slice -> A fragment (same-wave, no barrier) --
    #pragma unroll
    for (int t = 0; t < 2; ++t)
      #pragma unroll
      for (int r = 0; r < 8; ++r)
        Ps[wave][r + 8 * half][t * 16 + l15] = (f16)s[t][r];
    v16h pf = cat8(*(const v8h*)&Ps[wave][l15][8 * half],
                   *(const v8h*)&Ps[wave][l15][16 + 8 * half]);
    // ---- P(16x32) x V(32x32): masked keys have P==0, stale V is harmless ----
    #pragma unroll
    for (int t = 0; t < 2; ++t) {
      int dcol = t * 16 + l15;
      v16h vf = cat8(*(const v8h*)&Vs[dcol][8 * half],
                     *(const v8h*)&Vs[dcol][16 + 8 * half]);
      if (t == 0)
        acc0 = __builtin_amdgcn_wmma_f32_16x16x32_f16(false, pf, false, vf,
                                                      (short)0, acc0, false, false);
      else
        acc1 = __builtin_amdgcn_wmma_f32_16x16x32_f16(false, pf, false, vf,
                                                      (short)0, acc1, false, false);
    }
  }
  #pragma unroll
  for (int r = 0; r < 8; ++r) {
    int m = q0 + r + 8 * half;
    if (m < Lq) {
      float inv = 1.f / run_l[r];
      size_t o = ((size_t)b * Lq + m) * 256 + h * DH + l15;
      Out[o] = acc0[r] * inv;
      Out[o + 16] = acc1[r] * inv;
    }
  }
}

// -------------------- MS-deformable attention sampling ---------------------
// grid: B*S_TOT blocks, block 256 (8 waves = 8 heads; lane = channel d)
__global__ void msdeform_kernel(const float* __restrict__ val,
                                const float* __restrict__ off,
                                const float* __restrict__ awl,
                                const float* __restrict__ vr,
                                float* __restrict__ out) {
  int bq = blockIdx.x;
  int b = bq / S_TOT, q = bq - b * S_TOT;
  int h = threadIdx.x >> 5, d = threadIdx.x & 31;
  int lvlq, rq, cq, Hq, Wq, stq;
  level_of(q, lvlq, rq, cq, Hq, Wq, stq);

  const float* awp = awl + (size_t)bq * 128 + h * 16;
  float wgt[16]; float mx = -1e30f;
  #pragma unroll
  for (int i = 0; i < 16; ++i) { wgt[i] = awp[i]; mx = fmaxf(mx, wgt[i]); }
  float ssum = 0.f;
  #pragma unroll
  for (int i = 0; i < 16; ++i) { wgt[i] = __expf(wgt[i] - mx); ssum += wgt[i]; }
  float invs = 1.f / ssum;

  const float* offp = off + (size_t)bq * 256 + h * 32;
  const int Hs[4] = {64, 32, 16, 8}, Ws[4] = {64, 32, 16, 8};
  const int Sts[4] = {0, 4096, 5120, 5376};
  float base_x = (cq + 0.5f) / (vr[(b * 4 + lvlq) * 2 + 0] * (float)Wq);
  float base_y = (rq + 0.5f) / (vr[(b * 4 + lvlq) * 2 + 1] * (float)Hq);
  float o = 0.f;
  for (int l = 0; l < 4; ++l) {
    int Hl = Hs[l], Wl = Ws[l], st = Sts[l];
    float refx = base_x * vr[(b * 4 + l) * 2 + 0];
    float refy = base_y * vr[(b * 4 + l) * 2 + 1];
    #pragma unroll
    for (int p = 0; p < 4; ++p) {
      float ox = offp[(l * 4 + p) * 2 + 0], oy = offp[(l * 4 + p) * 2 + 1];
      float x = (refx + ox / (float)Wl) * (float)Wl - 0.5f;
      float y = (refy + oy / (float)Hl) * (float)Hl - 0.5f;
      float x0f = floorf(x), y0f = floorf(y);
      float wx = x - x0f, wy = y - y0f;
      int x0 = (int)x0f, y0 = (int)y0f;
      float s = 0.f;
      #pragma unroll
      for (int cy = 0; cy < 2; ++cy)
        #pragma unroll
        for (int cx = 0; cx < 2; ++cx) {
          int xi = x0 + cx, yi = y0 + cy;
          float cw = (cy ? wy : 1.f - wy) * (cx ? wx : 1.f - wx);
          bool valid = (xi >= 0) && (xi < Wl) && (yi >= 0) && (yi < Hl);
          int xc = imin(imax(xi, 0), Wl - 1), yc = imin(imax(yi, 0), Hl - 1);
          float sv = val[((size_t)b * S_TOT + st + yc * Wl + xc) * 256 + h * 32 + d];
          s += valid ? sv * cw : 0.f;
        }
      o += s * wgt[l * 4 + p] * invs;
    }
  }
  out[(size_t)bq * 256 + h * 32 + d] = o;
}

// -------------------- Residual add + LayerNorm (D=256) ---------------------
__global__ void add_ln_kernel(const float* __restrict__ a, const float* __restrict__ b,
                              const float* __restrict__ g, const float* __restrict__ beta,
                              float* __restrict__ out) {
  int row = blockIdx.x, lane = threadIdx.x;
  const float* pa = a + (size_t)row * 256;
  const float* pb = b + (size_t)row * 256;
  float x[8]; float s = 0.f, s2 = 0.f;
  #pragma unroll
  for (int i = 0; i < 8; ++i) {
    float v = pa[lane + i * 32] + pb[lane + i * 32];
    x[i] = v; s += v; s2 += v * v;
  }
  #pragma unroll
  for (int m = 16; m >= 1; m >>= 1) { s += __shfl_xor(s, m, 32); s2 += __shfl_xor(s2, m, 32); }
  float mean = s * (1.f / 256.f);
  float var = s2 * (1.f / 256.f) - mean * mean;
  float rinv = rsqrtf(var + 1e-5f);
  float* po = out + (size_t)row * 256;
  #pragma unroll
  for (int i = 0; i < 8; ++i) {
    int c = lane + i * 32;
    po[c] = (x[i] - mean) * rinv * g[c] + beta[c];
  }
}

// -------------------- elementwise helpers ----------------------------------
__global__ void add2_kernel(const float* a, const float* b, float* o, int n) {
  int i = blockIdx.x * 256 + threadIdx.x; if (i < n) o[i] = a[i] + b[i];
}
__global__ void posadd_kernel(const float* pos, const float* lvlemb, float* o, int n) {
  int i = blockIdx.x * 256 + threadIdx.x;
  if (i < n) {
    int d = i & 255, s = (i >> 8) % S_TOT;
    int lvl, r, c, H, W, st; level_of(s, lvl, r, c, H, W, st);
    o[i] = pos[i] + lvlemb[lvl * 256 + d];
  }
}
__global__ void fuse_kernel(const float* a, const float* b, const float* alpha, float* o, int n) {
  int i = blockIdx.x * 256 + threadIdx.x;
  if (i < n) { float al = alpha[0]; o[i] = al * a[i] + (1.f - al) * b[i]; }
}
__global__ void gather_kernel(const float* f, float* mem, int n) {
  int i = blockIdx.x * 256 + threadIdx.x;
  if (i < n) {
    int d = i & 255, pi = (i >> 8) % MEMPOOL, b = i / (MEMPOOL * 256);
    int s = (pi * S_TOT) / MEMPOOL;
    mem[i] = f[((size_t)b * S_TOT + s) * 256 + d];
  }
}
__global__ void bq_kernel(const float* qe, float* o, int n) {
  int i = blockIdx.x * 256 + threadIdx.x;
  if (i < n) { int d = i & 255, qi = (i >> 8) % NQ; o[i] = qe[qi * 256 + d]; }
}

// -------------------- host orchestration -----------------------------------
extern "C" void kernel_launch(void* const* d_in, const int* in_sizes, int n_in,
                              void* d_out, int out_size, void* d_ws, size_t ws_size,
                              hipStream_t stream) {
  (void)in_sizes; (void)n_in; (void)out_size; (void)ws_size;
  auto P = [&](int i) -> const float* { return (const float*)d_in[i]; };
  const float* src1 = P(0); const float* src2 = P(1); const float* pos = P(2);
  const float* vr = P(3); const float* lvlemb = P(4); const float* alpha = P(5);
  const float* qemb = P(6);
  const int E1 = 7, E2 = 103, CR = 199, DE = 235;

  const size_t NS = (size_t)BATCH * S_TOT;  // 21760 rows
  const size_t ND = NS * 256;
  float* w = (float*)d_ws; size_t ofs = 0;
  auto alloc = [&](size_t n) { float* p = w + ofs; ofs += n; return p; };
  float* POS = alloc(ND); float* M1 = alloc(ND); float* M2 = alloc(ND);
  float* F  = alloc(ND); float* QQ = alloc(ND); float* KP = alloc(ND);
  float* VP = alloc(ND); float* T0 = alloc(ND); float* T1 = alloc(ND);
  float* T3 = alloc(ND); float* T4 = alloc(ND);
  float* T2 = alloc(NS * 128);
  float* HF = alloc(NS * 1024);
  float* MEM = alloc((size_t)BATCH * MEMPOOL * 256);
  float* DQ  = alloc((size_t)BATCH * NQ * 256);

  auto gemm = [&](const float* A, const float* Wt, const float* bias, float* C,
                  int M, int N, int K, int relu) {
    dim3 g((M + 63) / 64, (N + 127) / 128);
    gemm_wmma_kernel<<<g, 256, 0, stream>>>(A, Wt, bias, C, M, N, K, relu);
  };
  auto addln = [&](const float* a, const float* b, const float* gg, const float* bb,
                   float* o, int rows) {
    add_ln_kernel<<<rows, 32, 0, stream>>>(a, b, gg, bb, o);
  };
  auto ew = [&](size_t n) { return dim3((unsigned)((n + 255) / 256)); };

  posadd_kernel<<<ew(ND), 256, 0, stream>>>(pos, lvlemb, POS, (int)ND);
  hipMemcpyAsync(M1, src1, ND * 4, hipMemcpyDeviceToDevice, stream);
  hipMemcpyAsync(M2, src2, ND * 4, hipMemcpyDeviceToDevice, stream);

  auto enc_layer = [&](int PB, float* M) {
    add2_kernel<<<ew(ND), 256, 0, stream>>>(M, POS, QQ, (int)ND);        // q = src+pos
    gemm(M,  P(PB + 7), P(PB + 6), T0, (int)NS, 256, 256, 0);            // value_proj
    gemm(QQ, P(PB + 3), P(PB + 2), T1, (int)NS, 256, 256, 0);            // offset
    gemm(QQ, P(PB + 1), P(PB + 0), T2, (int)NS, 128, 256, 0);            // aweight
    msdeform_kernel<<<(int)NS, 256, 0, stream>>>(T0, T1, T2, vr, T3);
    gemm(T3, P(PB + 5), P(PB + 4), T4, (int)NS, 256, 256, 0);            // out_proj
    addln(M, T4, P(PB + 13), P(PB + 12), M, (int)NS);                    // norm1
    gemm(M,  P(PB + 9), P(PB + 8), HF, (int)NS, 1024, 256, 1);           // lin1+relu
    gemm(HF, P(PB + 11), P(PB + 10), T4, (int)NS, 256, 1024, 0);         // lin2
    addln(M, T4, P(PB + 15), P(PB + 14), M, (int)NS);                    // norm2
  };
  for (int i = 0; i < 6; ++i) enc_layer(E1 + 16 * i, M1);
  for (int i = 0; i < 6; ++i) enc_layer(E2 + 16 * i, M2);

  fuse_kernel<<<ew(ND), 256, 0, stream>>>(M1, M2, alpha, F, (int)ND);

  auto mha = [&](const float* inb, const float* inw, const float* ob, const float* ow,
                 const float* Xq, const float* Xkv, int Lq, int Lk) {
    gemm(Xq,  inw,                 inb,       QQ, BATCH * Lq, 256, 256, 0);
    gemm(Xkv, inw + 256 * 256,     inb + 256, KP, BATCH * Lk, 256, 256, 0);
    gemm(Xkv, inw + 2 * 256 * 256, inb + 512, VP, BATCH * Lk, 256, 256, 0);
    attn_flash_kernel<<<dim3((Lq + 127) / 128, NHEADS, BATCH), 256, 0, stream>>>(
        QQ, KP, VP, T3, Lq, Lk);
    gemm(T3, ow, ob, T4, BATCH * Lq, 256, 256, 0);
  };
  auto cross_block = [&](int CB, float* Xq, const float* Xkv, int Lq, int Lk) {
    mha(P(CB + 4), P(CB + 5), P(CB + 6), P(CB + 7), Xq, Xkv, Lq, Lk);
    addln(Xq, T4, P(CB + 9), P(CB + 8), Xq, BATCH * Lq);                 // norm1
    gemm(Xq, P(CB + 1), P(CB + 0), HF, BATCH * Lq, 1024, 256, 1);        // ffn1+relu
    gemm(HF, P(CB + 3), P(CB + 2), T4, BATCH * Lq, 256, 1024, 0);        // ffn2
    addln(Xq, T4, P(CB + 11), P(CB + 10), Xq, BATCH * Lq);               // norm2
  };
  for (int i = 0; i < 3; ++i) cross_block(CR + 12 * i, F, M2, S_TOT, S_TOT);

  gather_kernel<<<ew((size_t)BATCH * MEMPOOL * 256), 256, 0, stream>>>(
      F, MEM, BATCH * MEMPOOL * 256);
  bq_kernel<<<ew((size_t)BATCH * NQ * 256), 256, 0, stream>>>(
      qemb, DQ, BATCH * NQ * 256);

  for (int i = 0; i < 6; ++i) {
    int DB = DE + 18 * i;
    mha(P(DB + 14), P(DB + 15), P(DB + 16), P(DB + 17), DQ, DQ, NQ, NQ); // self
    addln(DQ, T4, P(DB + 13), P(DB + 12), DQ, BATCH * NQ);               // norm1
    cross_block(DB, DQ, MEM, NQ, MEMPOOL);                               // cross
  }
  hipMemcpyAsync(d_out, DQ, (size_t)BATCH * NQ * 256 * 4,
                 hipMemcpyDeviceToDevice, stream);
}
